// DecoderNet_10196252360862
// MI455X (gfx1250) — matrix operand order repaired and checked by reading, not compiled, not run on hardware
//
#include <hip/hip_runtime.h>
#include <hip/hip_bf16.h>

// ---------------------------------------------------------------------------
// DecoderNet on MI455X (gfx1250, wave32, WMMA + async LDS DMA).
//
// Algebraic collapse: the 4-layer attention MLP has no activations, so
// score = [enc_out|h] @ (L1^T L2^T L3^T L4^T att_v) + const; the h-term is
// constant over T and cancels under softmax => attention weights and ctx are
// step-invariant. ctx is computed ONCE; ctx@W_ih[:,D:]^T is a per-batch
// constant; the embedded-target GEMM and the 32000-wide logits GEMM are
// batched. Only the 64x3072x1024 recurrent GEMM stays sequential (27 steps).
//
// Data movement: GLOBAL_LOAD_ASYNC_TO_LDS_B128 (ASYNCcnt) with 2-stage LDS
// double buffering — prefetch K-stage k+1 while v_wmma consumes stage k.
// Big GEMMs: v_wmma_f32_16x16x32_bf16.  Recurrent GEMM: v_wmma_f32_16x16x4_f32.
// ---------------------------------------------------------------------------

#define Bb   64
#define Tt   128
#define Ll   28
#define LM1  27
#define Hh   1024
#define Dd   1024
#define Vv   32000
#define H3   3072
#define HD   2048

typedef __bf16 v16bf __attribute__((ext_vector_type(16)));
typedef float  v8f   __attribute__((ext_vector_type(8)));
typedef float  v2f   __attribute__((ext_vector_type(2)));

__device__ __forceinline__ __bf16 f2bf(float f) {
  unsigned u = __builtin_bit_cast(unsigned, f);
  unsigned r = (u + 0x7FFFu + ((u >> 16) & 1u)) >> 16;   // round-to-nearest-even
  return __builtin_bit_cast(__bf16, (unsigned short)r);
}

// ---- CDNA5 async global->LDS DMA (per-lane 16B, tracked by ASYNCcnt) ------
// ISA 10.2: flat LDS address truncates to the wave-relative LDS byte offset,
// so a generic pointer to __shared__ gives the VDST value directly.
__device__ __forceinline__ void async_b128(void* lds, const void* gmem) {
  asm volatile("global_load_async_to_lds_b128 %0, %1, off"
               :: "v"((unsigned)(uintptr_t)lds), "v"(gmem)
               : "memory");
}
__device__ __forceinline__ void wait_async0() {
  asm volatile("s_wait_asynccnt 0x0" ::: "memory");
}

// ---------------- small precompute kernels ---------------------------------

// out[j] = sum_i M[i*ld + j] * v[i]   (transposed matvec for the u-chain)
__global__ void matvec_t_kernel(const float* __restrict__ M, int rows, int ld,
                                int ncols, const float* __restrict__ v,
                                float* __restrict__ out) {
  int j = blockIdx.x * blockDim.x + threadIdx.x;
  if (j >= ncols) return;
  float s = 0.f;
  for (int i = 0; i < rows; ++i) s += M[(size_t)i * ld + j] * v[i];
  out[j] = s;
}

// scores[b*T+t] = enc_out(b,t,:) . u      (one 128-thread block per row)
__global__ void scores_kernel(const float* __restrict__ enc,
                              const float* __restrict__ u,
                              float* __restrict__ scores) {
  int row = blockIdx.x;
  const float* e = enc + (size_t)row * Hh;
  float s = 0.f;
  for (int k = threadIdx.x; k < Hh; k += 128) s += e[k] * u[k];
  __shared__ float red[128];
  red[threadIdx.x] = s; __syncthreads();
  for (int off = 64; off > 0; off >>= 1) {
    if (threadIdx.x < off) red[threadIdx.x] += red[threadIdx.x + off];
    __syncthreads();
  }
  if (threadIdx.x == 0) scores[row] = red[0];
}

// in-place softmax over T=128 per batch row
__global__ void softmax_kernel(float* __restrict__ attn) {
  int b = blockIdx.x, t = threadIdx.x;
  __shared__ float red[128];
  float x = attn[b * Tt + t];
  red[t] = x; __syncthreads();
  for (int off = 64; off > 0; off >>= 1) {
    if (t < off) red[t] = fmaxf(red[t], red[t + off]);
    __syncthreads();
  }
  float m = red[0]; __syncthreads();
  float e = __expf(x - m);
  red[t] = e; __syncthreads();
  for (int off = 64; off > 0; off >>= 1) {
    if (t < off) red[t] += red[t + off];
    __syncthreads();
  }
  attn[b * Tt + t] = e / red[0];
}

// ctx[b,h] = sum_t attn[b,t] * enc_out[b,t,h]
__global__ void ctx_kernel(const float* __restrict__ enc,
                           const float* __restrict__ attn,
                           float* __restrict__ ctx) {
  int b = blockIdx.y;
  int h = blockIdx.x * 256 + threadIdx.x;
  float s = 0.f;
  for (int t = 0; t < Tt; ++t)
    s += attn[b * Tt + t] * enc[((size_t)b * Tt + t) * Hh + h];
  ctx[(size_t)b * Hh + h] = s;
}

// strided f32 -> bf16 (rows x cols out of a row-major matrix with leading dim ldin)
__global__ void convert_bf16_kernel(const float* __restrict__ in, int ldin,
                                    __bf16* __restrict__ out, int cols,
                                    long total) {
  long i = (long)blockIdx.x * blockDim.x + threadIdx.x;
  if (i >= total) return;
  long r = i / cols, c = i - r * cols;
  out[i] = f2bf(in[r * (long)ldin + c]);
}

// X[t*B+b, :] = bf16(emb[targets[b,t], :])   for t in [0, L-1)
__global__ void embed_kernel(const int* __restrict__ targets,
                             const float* __restrict__ emb,
                             __bf16* __restrict__ X) {
  int row = blockIdx.x;              // t*B + b
  int t = row / Bb, b = row - t * Bb;
  int tok = targets[b * Ll + t];
  const float* e = emb + (size_t)tok * Dd;
  __bf16* x = X + (size_t)row * Dd;
  for (int k = threadIdx.x; k < Dd; k += blockDim.x) x[k] = f2bf(e[k]);
}

// ---------------- bf16 WMMA GEMM:  C = A(MxK) @ Wb(NxK)^T (+bias) ----------
// block = 128 threads (4 waves), 64x64 tile; wave w -> rows [16w,16w+16),
// 4 accumulators of 16x16. Double-buffered LDS fed by async-to-LDS DMA.
// out_mode 0: C[m*N+n]; out_mode 1: logits remap m=t*B+b -> row (b*27+t).
#define LDSP 40
__global__ __launch_bounds__(128)
void gemm_bf16_wmma_kernel(const __bf16* __restrict__ A,
                           const __bf16* __restrict__ Wb,
                           const float* __restrict__ bias,
                           float* __restrict__ Cout,
                           int K, int N, int out_mode) {
  __shared__ __attribute__((aligned(16))) __bf16 As[2][64 * LDSP];
  __shared__ __attribute__((aligned(16))) __bf16 Bs[2][64 * LDSP];
  const int tid = threadIdx.x;
  const int wave = tid >> 5, lane = tid & 31;
  const int half = lane >> 4, nl = lane & 15;
  const int m0 = blockIdx.y * 64;
  const int n0 = blockIdx.x * 64;
  const int lr = tid >> 1;             // 0..63  (tile row to load)
  const int lc = (tid & 1) * 16;       // 0/16   (32B = 2 async b128 per matrix)

  const __bf16* gA = A  + (size_t)(m0 + lr) * K + lc;
  const __bf16* gB = Wb + (size_t)(n0 + lr) * K + lc;

  auto stage = [&](int buf, int k0) {
    async_b128(&As[buf][lr * LDSP + lc],     gA + k0);
    async_b128(&As[buf][lr * LDSP + lc + 8], gA + k0 + 8);
    async_b128(&Bs[buf][lr * LDSP + lc],     gB + k0);
    async_b128(&Bs[buf][lr * LDSP + lc + 8], gB + k0 + 8);
  };

  v8f acc[4] = {};
  stage(0, 0);
  wait_async0();
  __syncthreads();

  int buf = 0;
  for (int k0 = 0; k0 < K; k0 += 32) {
    if (k0 + 32 < K) stage(buf ^ 1, k0 + 32);   // prefetch next stage

    // fragments (contiguous k = 16*half + s for A and B); all DS loads hoisted before the WMMAs
    union Frag { v16bf v; uint4 q[2]; } af, bfr[4];
    const __bf16* Ab = &As[buf][(wave * 16 + nl) * LDSP];
    af.q[0] = *(const uint4*)(Ab + half * 16);
    af.q[1] = *(const uint4*)(Ab + half * 16 + 8);
#pragma unroll
    for (int nt = 0; nt < 4; ++nt) {
      const __bf16* Bp = &Bs[buf][(nt * 16 + nl) * LDSP + half * 16];
      bfr[nt].q[0] = *(const uint4*)(Bp);
      bfr[nt].q[1] = *(const uint4*)(Bp + 8);
    }
#pragma unroll
    for (int nt = 0; nt < 4; ++nt)
      acc[nt] = __builtin_amdgcn_wmma_f32_16x16x32_bf16(
          false, af.v, false, bfr[nt].v, (short)0, acc[nt], false, false);

    wait_async0();
    __syncthreads();
    buf ^= 1;
  }

  for (int nt = 0; nt < 4; ++nt) {
    int n = n0 + nt * 16 + nl;
    float bv = bias ? bias[n] : 0.f;
    for (int r = 0; r < 8; ++r) {
      int m = m0 + wave * 16 + half * 8 + r;   // C layout: M = half*8+r, N = nl
      float val = acc[nt][r] + bv;
      size_t off;
      if (out_mode == 0) {
        off = (size_t)m * N + n;
      } else {
        int t = m / Bb, b = m - t * Bb;        // m = t*B + b
        off = ((size_t)b * LM1 + t) * (size_t)N + n;
      }
      Cout[off] = val;
    }
  }
}

// ---------------- f32 WMMA GEMM (M=64 fixed, recurrent path) ---------------
// C(64xN) = A(64xK) @ W[:, koff:koff+K]^T + bias   via v_wmma_f32_16x16x4_f32
#define LDSPF 36
__global__ __launch_bounds__(128)
void gemm_f32_wmma_kernel(const float* __restrict__ A,
                          const float* __restrict__ W, int ldw, int koff,
                          const float* __restrict__ bias,
                          float* __restrict__ Cout, int K, int N) {
  __shared__ __attribute__((aligned(16))) float As[2][64 * LDSPF];
  __shared__ __attribute__((aligned(16))) float Ws[2][64 * LDSPF];
  const int tid = threadIdx.x;
  const int wave = tid >> 5, lane = tid & 31;
  const int half = lane >> 4, nl = lane & 15;
  const int n0 = blockIdx.x * 64;
  const int lr = tid >> 1;
  const int lc = (tid & 1) * 16;       // 16 floats = 4 async b128 per matrix

  const float* gA = A + (size_t)lr * K + lc;
  const float* gW = W + (size_t)(n0 + lr) * ldw + koff + lc;

  auto stage = [&](int buf, int k0) {
#pragma unroll
    for (int j = 0; j < 4; ++j) {
      async_b128(&As[buf][lr * LDSPF + lc + 4 * j], gA + k0 + 4 * j);
      async_b128(&Ws[buf][lr * LDSPF + lc + 4 * j], gW + k0 + 4 * j);
    }
  };

  v8f acc[4] = {};
  stage(0, 0);
  wait_async0();
  __syncthreads();

  int buf = 0;
  for (int k0 = 0; k0 < K; k0 += 32) {
    if (k0 + 32 < K) stage(buf ^ 1, k0 + 32);

#pragma unroll
    for (int kk = 0; kk < 32; kk += 4) {
      // f32 A 16x4: lanes0-15 -> K={kk,kk+1}, lanes16-31 -> K={kk+2,kk+3}
      float2 av = *(const float2*)(&As[buf][(wave * 16 + nl) * LDSPF + kk + 2 * half]);
      float2 bv[4];
#pragma unroll
      for (int nt = 0; nt < 4; ++nt)
        bv[nt] = *(const float2*)(&Ws[buf][(nt * 16 + nl) * LDSPF + kk + 2 * half]);
      v2f a; a.x = av.x; a.y = av.y;
#pragma unroll
      for (int nt = 0; nt < 4; ++nt) {
        v2f b; b.x = bv[nt].x; b.y = bv[nt].y;
        acc[nt] = __builtin_amdgcn_wmma_f32_16x16x4_f32(
            false, a, false, b, (short)0, acc[nt], false, false);
      }
    }

    wait_async0();
    __syncthreads();
    buf ^= 1;
  }

  for (int nt = 0; nt < 4; ++nt) {
    int n = n0 + nt * 16 + nl;
    float bv = bias ? bias[n] : 0.f;
    for (int r = 0; r < 8; ++r) {
      int m = wave * 16 + half * 8 + r;
      Cout[(size_t)m * N + n] = acc[nt][r] + bv;
    }
  }
}

// ---------------- GRU gate fusion ------------------------------------------
__global__ void gru_gate_kernel(const float* __restrict__ gi_x_t,
                                const float* __restrict__ gi_ctx,
                                const float* __restrict__ gh,
                                const float* __restrict__ h_in,
                                float* __restrict__ h_out,
                                __bf16* __restrict__ Hbf_t) {
  int idx = blockIdx.x * 256 + threadIdx.x;   // b*H + j
  int b = idx >> 10, j = idx & 1023;
  size_t o = (size_t)b * H3;
  float gir = gi_x_t[o + j]          + gi_ctx[o + j];
  float giz = gi_x_t[o + Hh + j]     + gi_ctx[o + Hh + j];
  float gin = gi_x_t[o + 2 * Hh + j] + gi_ctx[o + 2 * Hh + j];
  float ghr = gh[o + j], ghz = gh[o + Hh + j], ghn = gh[o + 2 * Hh + j];
  float r = 1.f / (1.f + __expf(-(gir + ghr)));
  float z = 1.f / (1.f + __expf(-(giz + ghz)));
  float n = tanhf(gin + r * ghn);
  float hp = h_in[idx];
  float hn = (1.f - z) * n + z * hp;
  h_out[idx] = hn;
  Hbf_t[idx] = f2bf(hn);
}

// ---------------- argmax over V per output row -----------------------------
__global__ void argmax_kernel(const float* __restrict__ logits,
                              float* __restrict__ preds) {
  int row = blockIdx.x;                      // b*27 + t (d_out row order)
  const float* p = logits + (size_t)row * Vv;
  float best = -3.4e38f; int bi = 0;
  for (int v = threadIdx.x; v < Vv; v += 256) {
    float x = p[v];
    if (x > best) { best = x; bi = v; }
  }
  __shared__ float bv[256]; __shared__ int bix[256];
  bv[threadIdx.x] = best; bix[threadIdx.x] = bi; __syncthreads();
  for (int off = 128; off > 0; off >>= 1) {
    if (threadIdx.x < off) {
      float ov = bv[threadIdx.x + off]; int oi = bix[threadIdx.x + off];
      if (ov > bv[threadIdx.x] || (ov == bv[threadIdx.x] && oi < bix[threadIdx.x])) {
        bv[threadIdx.x] = ov; bix[threadIdx.x] = oi;
      }
    }
    __syncthreads();
  }
  if (threadIdx.x == 0) preds[row] = (float)bix[0];
}

// ---------------------------------------------------------------------------
extern "C" void kernel_launch(void* const* d_in, const int* in_sizes, int n_in,
                              void* d_out, int out_size, void* d_ws, size_t ws_size,
                              hipStream_t stream) {
  const float* enc_h0  = (const float*)d_in[0];   // (1,B,H)
  const float* enc_out = (const float*)d_in[1];   // (B,T,H)
  const int*   targets = (const int*)d_in[2];     // (B,L)
  const float* emb     = (const float*)d_in[3];   // (V,D)
  const float* W_ih    = (const float*)d_in[4];   // (3H, H+D)
  const float* W_hh    = (const float*)d_in[5];   // (3H, H)
  const float* b_ih    = (const float*)d_in[6];
  const float* b_hh    = (const float*)d_in[7];
  const float* l1_w    = (const float*)d_in[8];   // (H,2H)
  const float* l2_w    = (const float*)d_in[10];  // (H,H)
  const float* l3_w    = (const float*)d_in[12];
  const float* l4_w    = (const float*)d_in[14];
  const float* att_v   = (const float*)d_in[16];  // (H)
  const float* out_w   = (const float*)d_in[17];  // (V,H)
  const float* out_b   = (const float*)d_in[18];
  // (biases l*_b cancel under softmax shift-invariance; not needed)

  float* logits = (float*)d_out;                              // (B,27,V)
  float* preds  = (float*)d_out + (size_t)Bb * LM1 * Vv;      // (B,27)

  // workspace carve (~102 MB)
  char* p = (char*)d_ws;
  auto take = [&](size_t bytes) {
    char* r = p;
    p += (bytes + 255) & ~(size_t)255;
    return r;
  };
  float*  u_a    = (float*)take(Hh * 4);
  float*  u_b    = (float*)take(Hh * 4);
  float*  u      = (float*)take(Hh * 4);
  float*  attn   = (float*)take((size_t)Bb * Tt * 4);
  float*  ctx    = (float*)take((size_t)Bb * Hh * 4);
  float*  gi_ctx = (float*)take((size_t)Bb * H3 * 4);
  float*  gh     = (float*)take((size_t)Bb * H3 * 4);
  float*  h_buf  = (float*)take((size_t)Bb * Hh * 4);
  float*  gi_x   = (float*)take((size_t)LM1 * Bb * H3 * 4);
  __bf16* Xbf    = (__bf16*)take((size_t)LM1 * Bb * Dd * 2);
  __bf16* Hbf    = (__bf16*)take((size_t)LM1 * Bb * Hh * 2);
  __bf16* Wxbf   = (__bf16*)take((size_t)H3 * Dd * 2);
  __bf16* Wobf   = (__bf16*)take((size_t)Vv * Hh * 2);

  // 1) u = L1[:, :H]^T (L2^T (L3^T (L4^T att_v)))  — attention collapses to
  //    a fixed H-vector; softmax shift removes all bias/h terms.
  matvec_t_kernel<<<Hh / 256, 256, 0, stream>>>(l4_w, Hh, Hh, Hh, att_v, u_a);
  matvec_t_kernel<<<Hh / 256, 256, 0, stream>>>(l3_w, Hh, Hh, Hh, u_a, u_b);
  matvec_t_kernel<<<Hh / 256, 256, 0, stream>>>(l2_w, Hh, Hh, Hh, u_b, u_a);
  matvec_t_kernel<<<Hh / 256, 256, 0, stream>>>(l1_w, Hh, HD, Hh, u_a, u);

  // 2) step-invariant attention + ctx
  scores_kernel<<<Bb * Tt, 128, 0, stream>>>(enc_out, u, attn);
  softmax_kernel<<<Bb, Tt, 0, stream>>>(attn);
  ctx_kernel<<<dim3(Hh / 256, Bb), 256, 0, stream>>>(enc_out, attn, ctx);

  // 3) bf16 weight staging (L2-resident afterwards)
  convert_bf16_kernel<<<(unsigned)(((long)H3 * Dd + 255) / 256), 256, 0, stream>>>(
      W_ih, HD, Wxbf, Dd, (long)H3 * Dd);
  convert_bf16_kernel<<<(unsigned)(((long)Vv * Hh + 255) / 256), 256, 0, stream>>>(
      out_w, Hh, Wobf, Hh, (long)Vv * Hh);

  // 4) batched embedding + input-projection GEMM: gi_x = Xbf @ Wxbf^T
  embed_kernel<<<LM1 * Bb, 256, 0, stream>>>(targets, emb, Xbf);
  gemm_bf16_wmma_kernel<<<dim3(H3 / 64, (LM1 * Bb) / 64), 128, 0, stream>>>(
      Xbf, Wxbf, nullptr, gi_x, Dd, H3, 0);

  // 5) gi_ctx = ctx @ W_ih[:, D:]^T + b_ih   (f32 WMMA, once)
  gemm_f32_wmma_kernel<<<H3 / 64, 128, 0, stream>>>(
      ctx, W_ih, HD, Dd, b_ih, gi_ctx, Hh, H3);

  // 6) sequential GRU: only gh = h @ W_hh^T + b_hh is on the critical path
  for (int t = 0; t < LM1; ++t) {
    const float* h_in = (t == 0) ? enc_h0 : h_buf;
    gemm_f32_wmma_kernel<<<H3 / 64, 128, 0, stream>>>(
        h_in, W_hh, Hh, 0, b_hh, gh, Hh, H3);
    gru_gate_kernel<<<(Bb * Hh) / 256, 256, 0, stream>>>(
        gi_x + (size_t)t * Bb * H3, gi_ctx, gh, h_in, h_buf,
        Hbf + (size_t)t * Bb * Hh);
  }

  // 7) all logits in one big bf16 WMMA GEMM (1728 x 32000 x 1024),
  //    written directly into d_out with (t,b)->(b,t) remap
  gemm_bf16_wmma_kernel<<<dim3(Vv / 64, (LM1 * Bb) / 64), 128, 0, stream>>>(
      Hbf, Wobf, out_b, logits, Hh, Vv, 1);

  // 8) argmax -> predictions (as float, matching d_out dtype)
  argmax_kernel<<<Bb * LM1, 256, 0, stream>>>(logits, preds);
}
